// HGNN_83485574300037
// MI455X (gfx1250) — matrix operand order repaired and checked by reading, not compiled, not run on hardware
//
#include <hip/hip_runtime.h>
#include <math.h>

// ---------------- problem constants ----------------
#define BB   16
#define NN   1024
#define FF   32
#define EE   4
#define NNZV 32768
#define HID  64
#define GG   64
#define ROWS (BB*NN)          // 16384
#define EPSV 1e-5f

typedef __attribute__((ext_vector_type(16))) __bf16 v16bf;
typedef __attribute__((ext_vector_type(8)))  float  v8f;

// ---------------- workspace layout (bytes) ----------------
// h0      : f32 [16384][64]                      4,194,304 B
// sums    : f32 [64] + [64] partial accumulators
// scale/shift : f32 [64] + [64]
// Wcomb   : f32 [4][64][64]  (= W1 @ Ws[c])
// bcomb   : f32 [4][64]      (= b1 @ Ws[c])
// HbT     : bf16 [b][c][g][n]   8,388,608 B  (H_ transposed, B-operand)
// M       : bf16 [b][c][j][k] 134,217,728 B  (mixed adjacency, A-operand)
#define OFF_H0     0ull
#define OFF_SUM    4194304ull
#define OFF_SUMSQ  (OFF_SUM + 256ull)
#define OFF_SCALE  (OFF_SUMSQ + 256ull)
#define OFF_SHIFT  (OFF_SCALE + 256ull)
#define OFF_WCOMB  (OFF_SHIFT + 256ull)
#define OFF_BCOMB  (OFF_WCOMB + 65536ull)
#define OFF_HBT    (OFF_BCOMB + 1024ull)          // 4,261,888 (16B aligned)
#define OFF_M      (OFF_HBT + 8388608ull)         // 12,650,496 (16B aligned)

// ---------------- bf16 helpers ----------------
__device__ __forceinline__ float bf2f(unsigned short h) {
    return __uint_as_float(((unsigned)h) << 16);
}
__device__ __forceinline__ unsigned short f2bf(float f) {
    unsigned u = __float_as_uint(f);
    u += 0x7FFFu + ((u >> 16) & 1u);              // round-to-nearest-even
    return (unsigned short)(u >> 16);
}
__device__ __forceinline__ void atomicAddBf16(unsigned short* p, float v) {
    size_t a = (size_t)p;
    unsigned* w = (unsigned*)(a & ~(size_t)3);
    const bool hi = (a & 2) != 0;
    unsigned old = *w, assumed;
    do {
        assumed = old;
        unsigned short h = hi ? (unsigned short)(assumed >> 16)
                              : (unsigned short)(assumed & 0xFFFFu);
        unsigned short nh = f2bf(bf2f(h) + v);
        unsigned nw = hi ? ((assumed & 0x0000FFFFu) | ((unsigned)nh << 16))
                         : ((assumed & 0xFFFF0000u) | (unsigned)nh);
        old = atomicCAS(w, assumed, nw);
    } while (old != assumed);
}

// ---------------- kernels ----------------
__global__ void k_zero4(uint4* __restrict__ p, size_t n4) {
    size_t i = (size_t)blockIdx.x * blockDim.x + threadIdx.x;
    size_t stride = (size_t)gridDim.x * blockDim.x;
    uint4 z = {0u, 0u, 0u, 0u};
    for (; i < n4; i += stride) p[i] = z;
}

__global__ void k_zero_f(float* __restrict__ p, int n) {
    int i = blockIdx.x * blockDim.x + threadIdx.x;
    if (i < n) p[i] = 0.f;
}

// h0 = X @ W0 + b0      [16384 x 64], K = 32
__global__ __launch_bounds__(256) void k_gemm1(const float* __restrict__ X,
                                               const float* __restrict__ W0,
                                               const float* __restrict__ b0,
                                               float* __restrict__ h0) {
    int idx = blockIdx.x * blockDim.x + threadIdx.x;   // [0, 16384*64)
    int r = idx >> 6, c = idx & 63;
    float acc = b0[c];
    const float* xr = X + (size_t)r * FF;
#pragma unroll
    for (int k = 0; k < FF; ++k) acc += xr[k] * W0[k * 64 + c];
    h0[idx] = acc;
}

// per-column partial sums of h0 and h0^2 (128 blocks x 128 rows)
__global__ __launch_bounds__(256) void k_stats(const float* __restrict__ h0,
                                               float* __restrict__ gsum,
                                               float* __restrict__ gsumsq) {
    __shared__ float s1[256], s2[256];
    int t = threadIdx.x;
    int c = t & 63, rg = t >> 6;
    int rbase = blockIdx.x * 128 + rg * 32;
    float s = 0.f, q = 0.f;
    for (int i = 0; i < 32; ++i) {
        float v = h0[(size_t)(rbase + i) * 64 + c];
        s += v; q += v * v;
    }
    s1[t] = s; s2[t] = q;
    __syncthreads();
    if (t < 64) {
        float ts = s1[t] + s1[t + 64] + s1[t + 128] + s1[t + 192];
        float tq = s2[t] + s2[t + 64] + s2[t + 128] + s2[t + 192];
        atomicAdd(&gsum[t], ts);
        atomicAdd(&gsumsq[t], tq);
    }
}

// finalize BN scale/shift; also bcomb[c][g] = sum_d b1[d] * Ws[c][d][g]
__global__ __launch_bounds__(256) void k_finalize(const float* __restrict__ gsum,
                                                  const float* __restrict__ gsumsq,
                                                  const float* __restrict__ gamma0,
                                                  const float* __restrict__ beta0,
                                                  const float* __restrict__ b1,
                                                  const float* __restrict__ Ws,
                                                  float* __restrict__ scale,
                                                  float* __restrict__ shift,
                                                  float* __restrict__ bcomb) {
    int t = threadIdx.x;
    if (t < 64) {
        float mu = gsum[t] * (1.0f / ROWS);
        float var = gsumsq[t] * (1.0f / ROWS) - mu * mu;
        float sc = gamma0[t] * rsqrtf(var + EPSV);
        scale[t] = sc;
        shift[t] = beta0[t] - mu * sc;
    }
    // bcomb: 256 values, t -> (c = t>>6, g = t&63)
    int c = t >> 6, g = t & 63;
    float acc = 0.f;
#pragma unroll 4
    for (int d = 0; d < 64; ++d) acc += b1[d] * Ws[((size_t)c * 64 + d) * 64 + g];
    bcomb[t] = acc;
}

// Wcomb[c][h][g] = sum_d W1[h][d] * Ws[c][d][g]
__global__ __launch_bounds__(256) void k_wcomb(const float* __restrict__ W1,
                                               const float* __restrict__ Ws,
                                               float* __restrict__ Wcomb) {
    int idx = blockIdx.x * blockDim.x + threadIdx.x;  // [0, 4*64*64)
    int c = idx >> 12, h = (idx >> 6) & 63, g = idx & 63;
    float acc = 0.f;
#pragma unroll 4
    for (int d = 0; d < 64; ++d)
        acc += W1[h * 64 + d] * Ws[((size_t)c * 64 + d) * 64 + g];
    Wcomb[idx] = acc;
}

// hn = ELU(BN(h0)); HbT[b][c][g][n] = bf16( hn . Wcomb[c][:,g] + bcomb[c][g] )
__global__ __launch_bounds__(256) void k_project(const float* __restrict__ h0,
                                                 const float* __restrict__ scale,
                                                 const float* __restrict__ shift,
                                                 const float* __restrict__ Wcomb,
                                                 const float* __restrict__ bcomb,
                                                 unsigned short* __restrict__ HbT) {
    __shared__ float hn[64 * 65];                 // 64 rows x 64 cols, padded
    int t = threadIdx.x;
    int rbase = blockIdx.x * 64;                  // 64 rows per block
    int b = rbase >> 10, n0 = rbase & 1023;
#pragma unroll
    for (int i = 0; i < 16; ++i) {
        int e = t + 256 * i;                      // [0, 4096)
        int rl = e >> 6, cc = e & 63;
        float x = h0[(size_t)(rbase + rl) * 64 + cc] * scale[cc] + shift[cc];
        hn[rl * 65 + cc] = (x > 0.f) ? x : (expf(x) - 1.0f);
    }
    __syncthreads();
    int c = t >> 6, g = t & 63;
    float wcol[64];
#pragma unroll 8
    for (int h = 0; h < 64; ++h) wcol[h] = Wcomb[((size_t)c << 12) + h * 64 + g];
    float bc = bcomb[c * 64 + g];
    unsigned short* dst = HbT + (((size_t)(b * 4 + c) * 64 + g) << 10) + n0;
    for (int rl = 0; rl < 64; ++rl) {
        float acc = bc;
        const float* row = &hn[rl * 65];
#pragma unroll 8
        for (int h = 0; h < 64; ++h) acc += row[h] * wcol[h];
        dst[rl] = f2bf(acc);
    }
}

// scatter: M[b][c][src][dst] += filt[c][e]   (bf16, CAS-based)
__global__ __launch_bounds__(256) void k_scatter(const int* __restrict__ esrc,
                                                 const int* __restrict__ edst,
                                                 const float* __restrict__ filt,
                                                 unsigned short* __restrict__ M) {
    size_t idx = (size_t)blockIdx.x * blockDim.x + threadIdx.x;
    if (idx >= (size_t)BB * EE * NNZV) return;
    int b = (int)(idx >> 17);                     // E*NNZ = 131072
    int e = (int)((idx >> 15) & 3);
    int src = esrc[idx], dst = edst[idx];
#pragma unroll
    for (int c = 0; c < EE; ++c) {
        float v = filt[c * EE + e];
        atomicAddBf16(M + (((size_t)(b * 4 + c) << 10) + src) * 1024 + dst, v);
    }
}

// Hs[b][c] = M[b][c] (1024x1024 bf16) @ H_[b][c] (1024x64 bf16) -> f32
// one block = (b,c, 128-row strip); 8 waves, each wave = 16 rows x 64 cols
__global__ __launch_bounds__(256) void k_bigmm(const unsigned short* __restrict__ M,
                                               const unsigned short* __restrict__ HbT,
                                               float* __restrict__ out) {
    __shared__ unsigned short lb[64 * 264];       // 64 g-rows x 256 k (+8 pad)
    const int t = threadIdx.x;
    const int wave = t >> 5, lane = t & 31;
    const int bc = blockIdx.x >> 3, jt = blockIdx.x & 7;
    const int half = lane >> 4, l15 = lane & 15;
    const int j = jt * 128 + wave * 16 + l15;     // A row this lane owns
    const size_t mrow = ((size_t)bc * 1024 + j) * 1024;
    const uint4* Hb4 = (const uint4*)(HbT + (size_t)bc * 64 * 1024);

    union { uint4 u[2]; v16bf v; } A, Bv;
    v8f acc[4] = {};

    for (int kb = 0; kb < 4; ++kb) {
        __syncthreads();
        // stage HbT[g=0..63][kb*256 .. +255] into LDS (32 KB)
#pragma unroll
        for (int i = 0; i < 8; ++i) {
            int q = t + 256 * i;                  // 2048 chunks of 8 bf16
            int g = q >> 5;
            int kk8 = (q & 31) * 8;
            uint4 d = Hb4[((size_t)g * 1024 + kb * 256 + kk8) >> 3];
            *(uint4*)&lb[g * 264 + kk8] = d;
        }
        __syncthreads();
        if (kb < 3)
            __builtin_prefetch((const void*)(M + mrow + (size_t)(kb + 1) * 256), 0, 0);
#pragma unroll
        for (int ks = 0; ks < 8; ++ks) {
            const int k0 = kb * 256 + ks * 32;
            // A: lane(row j) K = half*8+[0..7] and 16+half*8+[0..7]
            const uint4* ap = (const uint4*)(M + mrow + k0 + half * 8);
            A.u[0] = ap[0];
            A.u[1] = ap[2];                       // +16 elements
            const int klocal = ks * 32 + half * 8;
#pragma unroll
            for (int gt = 0; gt < 4; ++gt) {
                // B: lane(col g) K = half*8+[0..7] and 16+half*8+[0..7]
                const unsigned short* bp = &lb[(gt * 16 + l15) * 264 + klocal];
                Bv.u[0] = *(const uint4*)bp;
                Bv.u[1] = *(const uint4*)(bp + 16);
                acc[gt] = __builtin_amdgcn_wmma_f32_16x16x32_bf16(
                    false, A.v, false, Bv.v, (short)0, acc[gt], false, false);
            }
        }
    }
    // C/D layout: VGPR r -> row r + 8*half, col = lane&15
    const int jrow = jt * 128 + wave * 16 + 8 * half;
    float* ob = out + ((size_t)bc << 10) * 64;
#pragma unroll
    for (int gt = 0; gt < 4; ++gt)
#pragma unroll
        for (int r = 0; r < 8; ++r)
            ob[(size_t)(jrow + r) * 64 + gt * 16 + l15] = acc[gt][r];
}

// ---------------- launcher ----------------
extern "C" void kernel_launch(void* const* d_in, const int* in_sizes, int n_in,
                              void* d_out, int out_size, void* d_ws, size_t ws_size,
                              hipStream_t stream) {
    const float* X      = (const float*)d_in[0];
    const float* W0     = (const float*)d_in[1];
    const float* b0     = (const float*)d_in[2];
    const float* gamma0 = (const float*)d_in[3];
    const float* beta0  = (const float*)d_in[4];
    const float* W1     = (const float*)d_in[5];
    const float* b1     = (const float*)d_in[6];
    const float* Ws     = (const float*)d_in[7];
    const float* filt   = (const float*)d_in[8];
    const int*   esrc   = (const int*)d_in[9];
    const int*   edst   = (const int*)d_in[10];
    float* out = (float*)d_out;

    char* ws = (char*)d_ws;
    float*          h0     = (float*)(ws + OFF_H0);
    float*          gsum   = (float*)(ws + OFF_SUM);
    float*          gsumsq = (float*)(ws + OFF_SUMSQ);
    float*          scale  = (float*)(ws + OFF_SCALE);
    float*          shift  = (float*)(ws + OFF_SHIFT);
    float*          Wcomb  = (float*)(ws + OFF_WCOMB);
    float*          bcomb  = (float*)(ws + OFF_BCOMB);
    unsigned short* HbT    = (unsigned short*)(ws + OFF_HBT);
    unsigned short* M      = (unsigned short*)(ws + OFF_M);

    // zero mixed-adjacency (128 MB) and stat accumulators
    k_zero4<<<4096, 256, 0, stream>>>((uint4*)M, (size_t)(134217728ull / 16));
    k_zero_f<<<1, 128, 0, stream>>>(gsum, 128);

    // embedding MLP chain (tiny)
    k_gemm1<<<(ROWS * 64) / 256, 256, 0, stream>>>(X, W0, b0, h0);
    k_stats<<<128, 256, 0, stream>>>(h0, gsum, gsumsq);
    k_finalize<<<1, 256, 0, stream>>>(gsum, gsumsq, gamma0, beta0, b1, Ws,
                                      scale, shift, bcomb);
    k_wcomb<<<64, 256, 0, stream>>>(W1, Ws, Wcomb);
    k_project<<<ROWS / 64, 256, 0, stream>>>(h0, scale, shift, Wcomb, bcomb, HbT);

    // build filt-mixed dense adjacency in bf16
    k_scatter<<<(BB * EE * NNZV) / 256, 256, 0, stream>>>(esrc, edst, filt, M);

    // dominant batched GEMM on WMMA
    k_bigmm<<<BB * EE * 8, 256, 0, stream>>>(M, HbT, out);

    (void)in_sizes; (void)n_in; (void)out_size; (void)ws_size;
}
